// bi_Mamba2_13022340841846
// MI455X (gfx1250) — matrix operand order repaired and checked by reading, not compiled, not run on hardware
//
#include <hip/hip_runtime.h>
#include <hip/hip_bf16.h>

#define DM     1024
#define DINNER 2048
#define DSSM   2048
#define HEADS  32
#define PDIM   64
#define NSTATE 128
#define CHUNK_ 256
#define NCH    8
#define BATCH  2
#define SEQ    2048
#define CONVD  2304      /* DSSM + 2*NSTATE */
#define DIP    4384      /* 2*DINNER + 2*NSTATE + HEADS */
#define DCONV  4
#define EPSF   1e-5f
#define MROWS  (BATCH*SEQ)   /* 4096 */

typedef __attribute__((ext_vector_type(16))) __bf16 v16bf;
typedef __attribute__((ext_vector_type(2)))  __bf16 bf16x2;
typedef __attribute__((ext_vector_type(8)))  float  v8f;

// A-fragment K index pattern (16-bit A 16x32): v0..3 -> 2v, v4..7 -> 16+2(v-4), +8*half
__device__ __forceinline__ int apat(int v, int half) {
  return ((v & 3) << 1) + ((v & 4) ? 16 : 0) + (half << 3);
}
// B-fragment K index pattern (16-bit B 32x16): same as A: v0..3 -> 2v, v4..7 -> 16+2(v-4), +8*half; lane = column
__device__ __forceinline__ int bpat(int v, int half) {
  return ((v & 3) << 1) + ((v & 4) ? 16 : 0) + (half << 3);
}
__device__ __forceinline__ __bf16 tobf(float f) { return (__bf16)f; }

__device__ __forceinline__ v8f wmma_bf16(v16bf a, v16bf b, v8f c) {
  return __builtin_amdgcn_wmma_f32_16x16x32_bf16(false, a, false, b, (short)0, c,
                                                 false, false);
}

// ---------------------------------------------------------------------------
// Generic GEMM: C[M,N] = A[M,K] @ Bt[N,K]^T (+ bias), fp32 in/out, bf16 WMMA.
// Register-blocked: each wave owns a 64x32 output (4 m-tiles x 2 n-tiles),
// 8 WMMAs per K-step of 32 -> 2x less A traffic, 4x less B traffic than 16x16.
// ---------------------------------------------------------------------------
__global__ __launch_bounds__(256) void k_gemm_bf16(
    const float* __restrict__ A, const float* __restrict__ Bt,
    const float* __restrict__ bias, float* __restrict__ C,
    int M, int N, int K)
{
  const int lane  = threadIdx.x & 31;
  const int wid   = blockIdx.x * (blockDim.x >> 5) + (threadIdx.x >> 5);
  const int nblk  = N >> 5;                 // 32-col super-tiles
  const int total = (M >> 6) * nblk;        // 64-row super-tiles
  if (wid >= total) return;                 // wave-uniform guard, EXEC stays full
  const int tm = wid / nblk, tn = wid % nblk;
  const int r = lane & 15, half = lane >> 4;

  const float* arow[4];
  const float* brow[2];
#pragma unroll
  for (int mi = 0; mi < 4; ++mi) arow[mi] = A  + (size_t)(tm * 64 + mi * 16 + r) * K;
#pragma unroll
  for (int ni = 0; ni < 2; ++ni) brow[ni] = Bt + (size_t)(tn * 32 + ni * 16 + r) * K;

  v8f acc[4][2] = {};
  for (int k0 = 0; k0 < K; k0 += 32) {
    if (k0 + 32 < K) {
#pragma unroll
      for (int mi = 0; mi < 4; ++mi) __builtin_prefetch(arow[mi] + k0 + 32, 0, 1);
#pragma unroll
      for (int ni = 0; ni < 2; ++ni) __builtin_prefetch(brow[ni] + k0 + 32, 0, 1);
    }
    v16bf af[4], bfr[2];
#pragma unroll
    for (int mi = 0; mi < 4; ++mi)
#pragma unroll
      for (int v = 0; v < 8; ++v) {
        const float2 f = *(const float2*)(arow[mi] + k0 + apat(v, half));
        af[mi][2 * v] = tobf(f.x); af[mi][2 * v + 1] = tobf(f.y);
      }
#pragma unroll
    for (int ni = 0; ni < 2; ++ni)
#pragma unroll
      for (int v = 0; v < 8; ++v) {
        const float2 f = *(const float2*)(brow[ni] + k0 + bpat(v, half));
        bfr[ni][2 * v] = tobf(f.x); bfr[ni][2 * v + 1] = tobf(f.y);
      }
#pragma unroll
    for (int mi = 0; mi < 4; ++mi)
#pragma unroll
      for (int ni = 0; ni < 2; ++ni)
        acc[mi][ni] = wmma_bf16(af[mi], bfr[ni], acc[mi][ni]);
  }
#pragma unroll
  for (int ni = 0; ni < 2; ++ni) {
    const int col = tn * 32 + ni * 16 + r;
    const float bv = bias ? bias[col] : 0.f;
#pragma unroll
    for (int mi = 0; mi < 4; ++mi) {
      float* crow = C + (size_t)(tm * 64 + mi * 16 + half * 8) * N + col;
#pragma unroll
      for (int i = 0; i < 8; ++i) crow[(size_t)i * N] = acc[mi][ni][i] + bv;
    }
  }
}

// ---------------------------------------------------------------------------
// Causal depthwise conv (W=4) + SiLU on the xBC slice of zx (channel-flip aware)
// ---------------------------------------------------------------------------
__global__ void k_conv_silu(const float* __restrict__ zx, const float* __restrict__ w,
                            const float* __restrict__ bias, float* __restrict__ out,
                            int flip)
{
  int idx = blockIdx.x * blockDim.x + threadIdx.x;
  if (idx >= MROWS * CONVD) return;
  int c = idx % CONVD;
  int l = (idx / CONVD) % SEQ;
  int b = idx / (CONVD * SEQ);
  int j = DSSM + c;
  int src = flip ? (DIP - 1 - j) : j;
  const float* base = zx + (size_t)(b * SEQ) * DIP + src;
  float acc = bias[c];
#pragma unroll
  for (int i = 0; i < DCONV; ++i) {
    int ls = l - (DCONV - 1) + i;
    if (ls >= 0) acc += base[(size_t)(ls)*DIP] * w[c * DCONV + i];
  }
  out[idx] = acc / (1.f + __expf(-acc));      // silu
}

// ---------------------------------------------------------------------------
// dt = softplus(dt_raw + dt_bias)
// ---------------------------------------------------------------------------
__global__ void k_dt(const float* __restrict__ zx, const float* __restrict__ dt_bias,
                     float* __restrict__ dt, int flip)
{
  int idx = blockIdx.x * blockDim.x + threadIdx.x;
  if (idx >= MROWS * HEADS) return;
  int h = idx % HEADS;
  int b = idx / (HEADS * SEQ);
  int l = (idx / HEADS) % SEQ;
  int j = DSSM + CONVD + h;
  int src = flip ? (DIP - 1 - j) : j;
  float x = zx[(size_t)(b * SEQ + l) * DIP + src] + dt_bias[h];
  dt[idx] = (x > 20.f) ? x : log1pf(__expf(x));
}

// ---------------------------------------------------------------------------
// cs[b,c,h,l] = cumsum_l( dt * (-exp(A_log[h])) ) within each chunk
// ---------------------------------------------------------------------------
__global__ void k_cumsum(const float* __restrict__ dt, const float* __restrict__ A_log,
                         float* __restrict__ cs)
{
  int t = blockIdx.x * blockDim.x + threadIdx.x;
  if (t >= BATCH * NCH * HEADS) return;
  int h = t % HEADS;
  int c = (t / HEADS) % NCH;
  int b = t / (HEADS * NCH);
  float Ah = -__expf(A_log[h]);
  float acc = 0.f;
  float* out = cs + ((size_t)(b * NCH + c) * HEADS + h) * CHUNK_;
  const float* d = dt + ((size_t)(b * SEQ) + c * CHUNK_) * HEADS + h;
  for (int l = 0; l < CHUNK_; ++l) { acc += d[(size_t)l * HEADS] * Ah; out[l] = acc; }
}

// ---------------------------------------------------------------------------
// states[b,c,h,p,n] = sum_l xdt[l,p] * B[l,n] * exp(cs_last - cs[l])   (WMMA)
// ---------------------------------------------------------------------------
__global__ __launch_bounds__(256) void k_states(
    const float* __restrict__ xbc, const float* __restrict__ dt,
    const float* __restrict__ cs, float* __restrict__ states)
{
  const int bid = blockIdx.x;
  const int h = bid % HEADS, c = (bid / HEADS) % NCH, b = bid / (HEADS * NCH);
  __shared__ float s_dt[CHUNK_], s_dec[CHUNK_];
  {
    int l = threadIdx.x;
    const float* csrow = cs + ((size_t)(b * NCH + c) * HEADS + h) * CHUNK_;
    float last = csrow[CHUNK_ - 1];
    s_dec[l] = __expf(last - csrow[l]);
    s_dt[l]  = dt[((size_t)(b * SEQ) + c * CHUNK_ + l) * HEADS + h];
  }
  __syncthreads();
  const int lane = threadIdx.x & 31, wave = threadIdx.x >> 5;
  const int r = lane & 15, half = lane >> 4;
  const int pt = wave & 3, ntb = (wave >> 2) * 4;
  const int p = pt * 16 + r;
  const float* xb = xbc + ((size_t)(b * SEQ) + c * CHUNK_) * CONVD;
  v8f acc[4] = {};
  for (int k0 = 0; k0 < CHUNK_; k0 += 32) {
    v16bf a;
#pragma unroll
    for (int v = 0; v < 8; ++v) {
      int l0 = k0 + apat(v, half);
      a[2 * v]     = tobf(xb[(size_t)l0 * CONVD + h * PDIM + p] * s_dt[l0]);
      a[2 * v + 1] = tobf(xb[(size_t)(l0 + 1) * CONVD + h * PDIM + p] * s_dt[l0 + 1]);
    }
#pragma unroll
    for (int j = 0; j < 4; ++j) {
      int n = (ntb + j) * 16 + r;
      v16bf bb;
#pragma unroll
      for (int v = 0; v < 8; ++v) {
        int l0 = k0 + bpat(v, half);
        bb[2 * v]     = tobf(xb[(size_t)l0 * CONVD + DSSM + n] * s_dec[l0]);
        bb[2 * v + 1] = tobf(xb[(size_t)(l0 + 1) * CONVD + DSSM + n] * s_dec[l0 + 1]);
      }
      acc[j] = wmma_bf16(a, bb, acc[j]);
    }
  }
  float* st = states + ((size_t)(b * NCH + c) * HEADS + h) * PDIM * NSTATE;
#pragma unroll
  for (int j = 0; j < 4; ++j)
#pragma unroll
    for (int i = 0; i < 8; ++i)
      st[(size_t)(pt * 16 + half * 8 + i) * NSTATE + (ntb + j) * 16 + r] = acc[j][i];
}

// ---------------------------------------------------------------------------
// Inter-chunk scan (in place): states[c] <- state entering chunk c (hprev)
// ---------------------------------------------------------------------------
__global__ void k_scan(float* __restrict__ states, const float* __restrict__ cs)
{
  int t = blockIdx.x * blockDim.x + threadIdx.x;
  if (t >= BATCH * HEADS * PDIM * NSTATE) return;
  int n = t % NSTATE;
  int p = (t / NSTATE) % PDIM;
  int h = (t / (NSTATE * PDIM)) % HEADS;
  int b = t / (NSTATE * PDIM * HEADS);
  float hp = 0.f;
  for (int c = 0; c < NCH; ++c) {
    size_t idx = (((size_t)(b * NCH + c) * HEADS + h) * PDIM + p) * NSTATE + n;
    float s = states[idx];
    states[idx] = hp;
    float cd = __expf(cs[((size_t)(b * NCH + c) * HEADS + h) * CHUNK_ + CHUNK_ - 1]);
    hp = cd * hp + s;
  }
}

// ---------------------------------------------------------------------------
// Y = Yoff + Ydiag + D*x per (b,c,h).  16 waves, one 16-row l-tile each.
//   Yoff  = (C @ hprev^T) * exp(cs[l])                  (WMMA, K=n)
//   Ydiag = (mask*exp(cs_l - cs_s) * (C @ B^T)) @ xdt   (WMMA, G via LDS relayout)
// cs row staged via gfx1250 async global->LDS copy (ASYNCcnt).
// ---------------------------------------------------------------------------
__global__ __launch_bounds__(512) void k_ydiag(
    const float* __restrict__ xbc, const float* __restrict__ dt,
    const float* __restrict__ cs, const float* __restrict__ hprev,
    const float* __restrict__ Dp, float* __restrict__ Y)
{
  const int bid = blockIdx.x;
  const int h = bid % HEADS, c = (bid / HEADS) % NCH, b = bid / (HEADS * NCH);
  __shared__ __bf16 s_xdtT[PDIM][CHUNK_];  // transposed: [p][s], 32 KB
  __shared__ __bf16 s_G[16][16][32];       // 16 KB, one 16x32 slab per wave
  __shared__ float  s_cs[CHUNK_], s_ecs[CHUNK_];
  const float* xb = xbc + ((size_t)(b * SEQ) + c * CHUNK_) * CONVD;
  const float* csrow = cs + ((size_t)(b * NCH + c) * HEADS + h) * CHUNK_;

  // --- async DMA of the cs row into LDS (global_load_async_to_lds_b32) ---
  if (threadIdx.x < CHUNK_) {                       // waves 0..7: EXEC uniform
    unsigned lds_addr = (unsigned)(size_t)&s_cs[threadIdx.x];
    const float* gp = csrow + threadIdx.x;
    asm volatile("global_load_async_to_lds_b32 %0, %1, off"
                 :: "v"(lds_addr), "v"(gp) : "memory");
  }
  asm volatile("s_wait_asynccnt 0x0" ::: "memory");
  __syncthreads();

  for (int t = threadIdx.x; t < CHUNK_; t += blockDim.x)
    s_ecs[t] = __expf(s_cs[t]);
  for (int t = threadIdx.x; t < CHUNK_ * PDIM; t += blockDim.x) {
    int l = t / PDIM, p = t % PDIM;
    float dtl = dt[((size_t)(b * SEQ) + c * CHUNK_ + l) * HEADS + h];
    s_xdtT[p][l] = tobf(xb[(size_t)l * CONVD + h * PDIM + p] * dtl);
  }
  __syncthreads();
  const int lane = threadIdx.x & 31, wave = threadIdx.x >> 5;  // wave == l-tile
  const int r = lane & 15, half = lane >> 4;
  const int lt = wave;

  // cache 4 A-fragments of the C-matrix rows (K = n = 128)
  const float* crow = xb + (size_t)(lt * 16 + r) * CONVD + (DSSM + NSTATE);
  v16bf afr[4];
#pragma unroll
  for (int kn = 0; kn < 4; ++kn)
#pragma unroll
    for (int v = 0; v < 8; ++v) {
      const float2 f = *(const float2*)(crow + kn * 32 + apat(v, half));
      afr[kn][2 * v] = tobf(f.x); afr[kn][2 * v + 1] = tobf(f.y);
    }

  v8f acc[4] = {};
  // ---- Yoff
  const float* hp = hprev + ((size_t)(b * NCH + c) * HEADS + h) * PDIM * NSTATE;
#pragma unroll
  for (int kn = 0; kn < 4; ++kn)
#pragma unroll
    for (int ptj = 0; ptj < 4; ++ptj) {
      v16bf bb;
#pragma unroll
      for (int v = 0; v < 8; ++v) {
        int n0 = kn * 32 + bpat(v, half);
        const float2 f = *(const float2*)(hp + (size_t)(ptj * 16 + r) * NSTATE + n0);
        bb[2 * v] = tobf(f.x); bb[2 * v + 1] = tobf(f.y);
      }
      acc[ptj] = wmma_bf16(afr[kn], bb, acc[ptj]);
    }
#pragma unroll
  for (int ptj = 0; ptj < 4; ++ptj)
#pragma unroll
    for (int i = 0; i < 8; ++i)
      acc[ptj][i] *= s_ecs[lt * 16 + half * 8 + i];

  // ---- causal diagonal part
  const int smax = (lt * 16 + 15) / 32;
  for (int sb = 0; sb <= smax; ++sb) {
    const int S = sb * 32;
    v8f g0 = {}, g1 = {};
#pragma unroll
    for (int kn = 0; kn < 4; ++kn) {
      v16bf b0, b1;
#pragma unroll
      for (int v = 0; v < 8; ++v) {
        int n0 = kn * 32 + bpat(v, half);
        const float2 f0 = *(const float2*)(xb + (size_t)(S + r) * CONVD + DSSM + n0);
        const float2 f1 = *(const float2*)(xb + (size_t)(S + 16 + r) * CONVD + DSSM + n0);
        b0[2 * v] = tobf(f0.x); b0[2 * v + 1] = tobf(f0.y);
        b1[2 * v] = tobf(f1.x); b1[2 * v + 1] = tobf(f1.y);
      }
      g0 = wmma_bf16(afr[kn], b0, g0);
      g1 = wmma_bf16(afr[kn], b1, g1);
    }
    // mask + exp(cs_l - cs_s), stash G tile row-major in this wave's LDS slab
#pragma unroll
    for (int i = 0; i < 8; ++i) {
      int lrow = lt * 16 + half * 8 + i;
      int s0 = S + r, s1 = S + 16 + r;
      float v0 = (lrow >= s0) ? g0[i] * __expf(s_cs[lrow] - s_cs[s0]) : 0.f;
      float v1 = (lrow >= s1) ? g1[i] * __expf(s_cs[lrow] - s_cs[s1]) : 0.f;
      s_G[wave][half * 8 + i][r]      = tobf(v0);
      s_G[wave][half * 8 + i][16 + r] = tobf(v1);
    }
    // reload G as an A-fragment (same-wave DS dependency; K-pairs contiguous)
    v16bf a2;
#pragma unroll
    for (int v = 0; v < 8; ++v) {
      int k = apat(v, half);
      const bf16x2 g = *(const bf16x2*)&s_G[wave][r][k];
      a2[2 * v] = g[0]; a2[2 * v + 1] = g[1];
    }
#pragma unroll
    for (int ptj = 0; ptj < 4; ++ptj) {
      const __bf16* xr = &s_xdtT[ptj * 16 + r][0];
      v16bf bb;
#pragma unroll
      for (int v = 0; v < 8; ++v) {
        const bf16x2 t = *(const bf16x2*)(xr + S + bpat(v, half));
        bb[2 * v] = t[0]; bb[2 * v + 1] = t[1];
      }
      acc[ptj] = wmma_bf16(a2, bb, acc[ptj]);
    }
  }

  // ---- + D*x, store
  const float Dh = Dp[h];
  float* yo = Y + ((size_t)(b * SEQ) + c * CHUNK_) * DSSM + h * PDIM;
#pragma unroll
  for (int ptj = 0; ptj < 4; ++ptj)
#pragma unroll
    for (int i = 0; i < 8; ++i) {
      int lrow = lt * 16 + half * 8 + i;
      int p = ptj * 16 + r;
      float xv = xb[(size_t)lrow * CONVD + h * PDIM + p];
      yo[(size_t)lrow * DSSM + p] = acc[ptj][i] + Dh * xv;
    }
}

// ---------------------------------------------------------------------------
// y = y * silu(z); y = y * rsqrt(mean(y^2)+eps) * norm_w    (one block per row)
// ---------------------------------------------------------------------------
__global__ __launch_bounds__(256) void k_gate_norm(
    const float* __restrict__ zx, const float* __restrict__ nw,
    float* __restrict__ Y, int flip)
{
  const int bl = blockIdx.x;
  float* yrow = Y + (size_t)bl * DSSM;
  const float* zrow = zx + (size_t)bl * DIP;
  __shared__ float red[256];
  float vals[DSSM / 256];
  float part = 0.f;
#pragma unroll
  for (int q = 0; q < DSSM / 256; ++q) {
    int f = q * 256 + threadIdx.x;
    int src = flip ? (DIP - 1 - f) : f;
    float z = zrow[src];
    float yv = yrow[f] * (z / (1.f + __expf(-z)));
    vals[q] = yv;
    part += yv * yv;
  }
  red[threadIdx.x] = part;
  __syncthreads();
  for (int s = 128; s > 0; s >>= 1) {
    if (threadIdx.x < s) red[threadIdx.x] += red[threadIdx.x + s];
    __syncthreads();
  }
  float scale = rsqrtf(red[0] / (float)DSSM + EPSF);
#pragma unroll
  for (int q = 0; q < DSSM / 256; ++q) {
    int f = q * 256 + threadIdx.x;
    yrow[f] = vals[q] * scale * nw[f];
  }
}

__global__ void k_combine(const float* __restrict__ a, const float* __restrict__ b,
                          const float* __restrict__ w1p, const float* __restrict__ w2p,
                          float* __restrict__ out)
{
  int i = blockIdx.x * blockDim.x + threadIdx.x;
  if (i >= MROWS * DM) return;
  out[i] = w1p[0] * a[i] + w2p[0] * b[i];
}

// ---------------------------------------------------------------------------
extern "C" void kernel_launch(void* const* d_in, const int* in_sizes, int n_in,
                              void* d_out, int out_size, void* d_ws, size_t ws_size,
                              hipStream_t stream)
{
  const float* u      = (const float*)d_in[0];
  const float* in_w   = (const float*)d_in[1];
  const float* convw[2] = {(const float*)d_in[2], (const float*)d_in[4]};
  const float* convb[2] = {(const float*)d_in[3], (const float*)d_in[5]};
  const float* dtb[2]   = {(const float*)d_in[6], (const float*)d_in[7]};
  const float* alog[2]  = {(const float*)d_in[8], (const float*)d_in[9]};
  const float* Dp[2]    = {(const float*)d_in[10], (const float*)d_in[11]};
  const float* normw  = (const float*)d_in[12];
  const float* outw   = (const float*)d_in[13];
  const float* endw   = (const float*)d_in[14];
  const float* endb   = (const float*)d_in[15];
  const float* w1     = (const float*)d_in[16];
  const float* w2     = (const float*)d_in[17];

  float* ws = (float*)d_ws;
  size_t o = 0;
  float* zx     = ws + o; o += (size_t)MROWS * DIP;
  float* xbc    = ws + o; o += (size_t)MROWS * CONVD;
  float* dtv    = ws + o; o += (size_t)MROWS * HEADS;
  float* csv    = ws + o; o += (size_t)BATCH * NCH * HEADS * CHUNK_;
  float* states = ws + o; o += (size_t)BATCH * NCH * HEADS * PDIM * NSTATE;
  float* Yb     = ws + o; o += (size_t)MROWS * DSSM;
  float* outf   = ws + o; o += (size_t)MROWS * DM;
  float* outb   = ws + o; o += (size_t)MROWS * DM;
  float* comb   = Yb;   // reuse: Y no longer needed after out_proj GEMMs
  float* outs[2] = {outf, outb};

  // in_proj: zx = u @ in_proj_w^T
  {
    int waves = (MROWS / 64) * (DIP / 32);
    k_gemm_bf16<<<(waves + 7) / 8, 256, 0, stream>>>(u, in_w, nullptr, zx,
                                                     MROWS, DIP, DM);
  }

  for (int br = 0; br < 2; ++br) {
    const int flip = br;
    k_conv_silu<<<(MROWS * CONVD + 255) / 256, 256, 0, stream>>>(
        zx, convw[br], convb[br], xbc, flip);
    k_dt<<<(MROWS * HEADS + 255) / 256, 256, 0, stream>>>(zx, dtb[br], dtv, flip);
    k_cumsum<<<2, 256, 0, stream>>>(dtv, alog[br], csv);
    k_states<<<BATCH * NCH * HEADS, 256, 0, stream>>>(xbc, dtv, csv, states);
    k_scan<<<(BATCH * HEADS * PDIM * NSTATE + 255) / 256, 256, 0, stream>>>(states, csv);
    k_ydiag<<<BATCH * NCH * HEADS, 512, 0, stream>>>(xbc, dtv, csv, states,
                                                     Dp[br], Yb);
    k_gate_norm<<<MROWS, 256, 0, stream>>>(zx, normw, Yb, flip);
    int waves = (MROWS / 64) * (DM / 32);
    k_gemm_bf16<<<(waves + 7) / 8, 256, 0, stream>>>(Yb, outw, nullptr, outs[br],
                                                     MROWS, DM, DINNER);
  }

  k_combine<<<(MROWS * DM + 255) / 256, 256, 0, stream>>>(outf, outb, w1, w2, comb);
  {
    int waves = (MROWS / 64) * (DM / 32);
    k_gemm_bf16<<<(waves + 7) / 8, 256, 0, stream>>>(comb, endw, endb,
                                                     (float*)d_out, MROWS, DM, DM);
  }
  (void)in_sizes; (void)n_in; (void)out_size; (void)ws_size;
}
